// EventDrivenAttention_22703197127225
// MI455X (gfx1250) — matrix operand order repaired and checked by reading, not compiled, not run on hardware
//
#include <hip/hip_runtime.h>
#include <hip/hip_bf16.h>

#define S_LEN   2048
#define D_MODEL 1024
#define N_HEADS 16
#define HDIM    64

typedef __attribute__((ext_vector_type(16))) __bf16 v16bf;
typedef __attribute__((ext_vector_type(8)))  __bf16 bf16x8;
typedef __attribute__((ext_vector_type(4)))  __bf16 bf16x4;
typedef __attribute__((ext_vector_type(8)))  float  v8f;
typedef __attribute__((ext_vector_type(4)))  float  f32x4;

__device__ __forceinline__ v16bf cat8(bf16x8 lo, bf16x8 hi) {
  return __builtin_shufflevector(lo, hi, 0,1,2,3,4,5,6,7,8,9,10,11,12,13,14,15);
}
__device__ __forceinline__ v8f wmma_bf16(v16bf a, v16bf b, v8f c) {
  return __builtin_amdgcn_wmma_f32_16x16x32_bf16(false, a, false, b, (short)0, c,
                                                 false, false);
}

// ---------------------------------------------------------------------------
// K1: qk[s][n] = sum_d x[s][d]*w[n][d] + bias[n]  (n in [0,2048): Q rows then K
// rows of in_proj_w), stored bf16. NT GEMM. Block tile 128x128, kk-chunk 64,
// 8 waves in 4(m) x 2(n) grid, each wave 32x64 (2x4 wmma fragments).
// ---------------------------------------------------------------------------
__global__ __launch_bounds__(256) void qk_proj_kernel(
    const float* __restrict__ x, const float* __restrict__ w,
    const float* __restrict__ bias, __bf16* __restrict__ qk) {
  __shared__ __align__(16) __bf16 aT[128][72];  // [m][k], 144B row (16B-aligned)
  __shared__ __align__(16) __bf16 bT[128][72];  // [n][k]
  const int tid = threadIdx.x;
  const int wave = tid >> 5, lane = tid & 31;
  const int half = lane >> 4, lrow = lane & 15;
  const int m0 = blockIdx.x * 128, n0 = blockIdx.y * 128;
  const int wm = (wave & 3) * 32, wn = (wave >> 2) * 64;

  v8f acc[2][4] = {};
  for (int kk = 0; kk < D_MODEL; kk += 64) {
    __syncthreads();
    #pragma unroll
    for (int i = 0; i < 8; ++i) {                 // A: 128 x 64 f32 -> bf16
      int ch = tid + i * 256;
      int r = ch >> 4, c = (ch & 15) << 2;
      f32x4 v = *(const f32x4*)&x[(m0 + r) * D_MODEL + kk + c];
      *(bf16x4*)&aT[r][c] = __builtin_convertvector(v, bf16x4);
    }
    #pragma unroll
    for (int i = 0; i < 8; ++i) {                 // B: 128 x 64 f32 -> bf16
      int ch = tid + i * 256;
      int r = ch >> 4, c = (ch & 15) << 2;
      f32x4 v = *(const f32x4*)&w[(n0 + r) * D_MODEL + kk + c];
      *(bf16x4*)&bT[r][c] = __builtin_convertvector(v, bf16x4);
    }
    __syncthreads();
    #pragma unroll
    for (int ks = 0; ks < 64; ks += 32) {
      v16bf af[2], bfr[4];
      #pragma unroll
      for (int mi = 0; mi < 2; ++mi) {
        const __bf16* p = &aT[wm + mi * 16 + lrow][ks + half * 8];
        af[mi] = cat8(*(const bf16x8*)p, *(const bf16x8*)(p + 16));
      }
      #pragma unroll
      for (int ni = 0; ni < 4; ++ni) {
        const __bf16* p = &bT[wn + ni * 16 + lrow][ks + half * 8];
        bfr[ni] = cat8(*(const bf16x8*)p, *(const bf16x8*)(p + 16));
      }
      #pragma unroll
      for (int mi = 0; mi < 2; ++mi)
        #pragma unroll
        for (int ni = 0; ni < 4; ++ni)
          acc[mi][ni] = wmma_bf16(af[mi], bfr[ni], acc[mi][ni]);
    }
  }
  #pragma unroll
  for (int mi = 0; mi < 2; ++mi)
    #pragma unroll
    for (int ni = 0; ni < 4; ++ni) {
      int col = n0 + wn + ni * 16 + lrow;
      float bb = bias[col];
      #pragma unroll
      for (int i = 0; i < 8; ++i) {
        int row = m0 + wm + mi * 16 + i + half * 8;
        qk[row * 2048 + col] = (__bf16)(acc[mi][ni][i] + bb);
      }
    }
}

// ---------------------------------------------------------------------------
// K2: fused scores -> softmax -> head-avg -> prox -> softmax -> biased.
// One block per 16-query tile; 8 waves; each wave owns a 16x256 score chunk in
// 16 v8f register accumulators. Head-average kept in 128KB LDS (<320KB/WGP).
// ---------------------------------------------------------------------------
__global__ __launch_bounds__(256) void attn_kernel(
    const __bf16* __restrict__ qk, const int* __restrict__ evt,
    float* __restrict__ biased) {
  __shared__ float avg[16][S_LEN];  // 128 KB
  __shared__ float red[16][16];
  __shared__ float rowmax[16];
  __shared__ float rowsum[16];

  const int tid = threadIdx.x;
  const int wave = tid >> 5, lane = tid & 31;
  const int half = lane >> 4, lrow = lane & 15;
  const int q0 = blockIdx.x * 16;
  const int c0 = wave * 256;  // this wave's key-column base

  {
    f32x4 z = {0.f, 0.f, 0.f, 0.f};
    #pragma unroll
    for (int i = 0; i < 32; ++i)
      *(f32x4*)(&avg[0][0] + (tid + i * 256) * 4) = z;
  }
  __syncthreads();

  for (int h = 0; h < N_HEADS; ++h) {
    const __bf16* qb = qk + (q0 + lrow) * 2048 + h * HDIM;
    v16bf qa0 = cat8(*(const bf16x8*)(qb + half * 8),
                     *(const bf16x8*)(qb + half * 8 + 16));
    v16bf qa1 = cat8(*(const bf16x8*)(qb + 32 + half * 8),
                     *(const bf16x8*)(qb + 32 + half * 8 + 16));
    v8f acc[16] = {};
    #pragma unroll
    for (int t = 0; t < 16; ++t) {
      const __bf16* kb = qk + (c0 + t * 16 + lrow) * 2048 + 1024 + h * HDIM + half * 8;
      v16bf kb0 = cat8(*(const bf16x8*)(kb),      *(const bf16x8*)(kb + 16));
      v16bf kb1 = cat8(*(const bf16x8*)(kb + 32), *(const bf16x8*)(kb + 48));
      acc[t] = wmma_bf16(qa0, kb0, acc[t]);
      acc[t] = wmma_bf16(qa1, kb1, acc[t]);
    }
    // scale by 1/sqrt(64), per-lane row-max partials
    float pm[8];
    #pragma unroll
    for (int i = 0; i < 8; ++i) pm[i] = -3.0e38f;
    #pragma unroll
    for (int t = 0; t < 16; ++t)
      #pragma unroll
      for (int i = 0; i < 8; ++i) {
        float v = acc[t][i] * 0.125f;
        acc[t][i] = v;
        pm[i] = fmaxf(pm[i], v);
      }
    #pragma unroll
    for (int i = 0; i < 8; ++i)
      #pragma unroll
      for (int m = 1; m < 16; m <<= 1)
        pm[i] = fmaxf(pm[i], __shfl_xor(pm[i], m, 32));
    if (lrow == 0) {
      #pragma unroll
      for (int i = 0; i < 8; ++i) red[i + half * 8][wave] = pm[i];
    }
    __syncthreads();
    if (tid < 16) {
      float m = -3.0e38f;
      for (int j = 0; j < 8; ++j) m = fmaxf(m, red[tid][j]);
      rowmax[tid] = m;
    }
    __syncthreads();
    float ps[8];
    #pragma unroll
    for (int i = 0; i < 8; ++i) ps[i] = 0.f;
    #pragma unroll
    for (int t = 0; t < 16; ++t)
      #pragma unroll
      for (int i = 0; i < 8; ++i) {
        float e = __expf(acc[t][i] - rowmax[i + half * 8]);
        acc[t][i] = e;
        ps[i] += e;
      }
    #pragma unroll
    for (int i = 0; i < 8; ++i)
      #pragma unroll
      for (int m = 1; m < 16; m <<= 1)
        ps[i] += __shfl_xor(ps[i], m, 32);
    if (lrow == 0) {
      #pragma unroll
      for (int i = 0; i < 8; ++i) red[i + half * 8][wave] = ps[i];
    }
    __syncthreads();
    if (tid < 16) {
      float s = 0.f;
      for (int j = 0; j < 8; ++j) s += red[tid][j];
      rowsum[tid] = s;
    }
    __syncthreads();
    #pragma unroll
    for (int i = 0; i < 8; ++i) {
      int r = i + half * 8;
      float inv = 1.f / (16.f * rowsum[r]);
      #pragma unroll
      for (int t = 0; t < 16; ++t)
        avg[r][c0 + t * 16 + lrow] += acc[t][i] * inv;
    }
    __syncthreads();
  }

  // prox weighting (row scalar) + second softmax + write `biased`
  const int ev = evt[0];
  const int row = tid >> 4;
  const int seg = tid & 15;
  const int gq = q0 + row;
  const int dlt = gq - ev;
  const float p = __expf(-(float)(dlt * dlt));  // EVENT_STRENGTH = 1
  float* rp = avg[row];
  float lmax = -3.0e38f;
  #pragma unroll 8
  for (int j = 0; j < 128; ++j) lmax = fmaxf(lmax, rp[seg + (j << 4)] * p);
  red[row][seg] = lmax;
  __syncthreads();
  if (tid < 16) {
    float m = -3.0e38f;
    for (int j = 0; j < 16; ++j) m = fmaxf(m, red[tid][j]);
    rowmax[tid] = m;
  }
  __syncthreads();
  const float M = rowmax[row];
  float lsum = 0.f;
  #pragma unroll 8
  for (int j = 0; j < 128; ++j) {
    int c = seg + (j << 4);
    float e = __expf(rp[c] * p - M);
    rp[c] = e;
    lsum += e;
  }
  red[row][seg] = lsum;
  __syncthreads();
  if (tid < 16) {
    float s = 0.f;
    for (int j = 0; j < 16; ++j) s += red[tid][j];
    rowsum[tid] = s;
  }
  __syncthreads();
  const float inv = 1.f / rowsum[row];
  #pragma unroll 8
  for (int j = 0; j < 128; ++j) {
    int c = seg + (j << 4);
    biased[gq * 2048 + c] = rp[c] * inv;
  }
}

// ---------------------------------------------------------------------------
// K3: out[q][d] = sum_k biased[q][k] * x[k][d]. NN GEMM; B (x) transposed into
// LDS during f32->bf16 staging so fragments are contiguous-in-K per lane.
// Same 128x128 / kk-chunk-64 / 32x64-per-wave tiling as K1.
// ---------------------------------------------------------------------------
__global__ __launch_bounds__(256) void out_gemm_kernel(
    const float* __restrict__ biased, const float* __restrict__ x,
    float* __restrict__ out) {
  __shared__ __align__(16) __bf16 aT[128][72];  // [q][k]
  __shared__ __align__(16) __bf16 bT[128][72];  // [d][k] (transposed x)
  const int tid = threadIdx.x;
  const int wave = tid >> 5, lane = tid & 31;
  const int half = lane >> 4, lrow = lane & 15;
  const int m0 = blockIdx.x * 128, n0 = blockIdx.y * 128;
  const int wm = (wave & 3) * 32, wn = (wave >> 2) * 64;

  v8f acc[2][4] = {};
  for (int kk = 0; kk < S_LEN; kk += 64) {
    __syncthreads();
    #pragma unroll
    for (int i = 0; i < 8; ++i) {                 // A: 128 q x 64 k
      int ch = tid + i * 256;
      int r = ch >> 4, c = (ch & 15) << 2;
      f32x4 v = *(const f32x4*)&biased[(m0 + r) * 2048 + kk + c];
      *(bf16x4*)&aT[r][c] = __builtin_convertvector(v, bf16x4);
    }
    #pragma unroll
    for (int i = 0; i < 8; ++i) {                 // B: 64 k x 128 d, transpose
      int ch = tid + i * 256;
      int k = ch >> 5, n4 = (ch & 31) << 2;
      f32x4 v = *(const f32x4*)&x[(kk + k) * D_MODEL + n0 + n4];
      bf16x4 b4 = __builtin_convertvector(v, bf16x4);
      bT[n4 + 0][k] = b4[0];
      bT[n4 + 1][k] = b4[1];
      bT[n4 + 2][k] = b4[2];
      bT[n4 + 3][k] = b4[3];
    }
    __syncthreads();
    #pragma unroll
    for (int ks = 0; ks < 64; ks += 32) {
      v16bf af[2], bfr[4];
      #pragma unroll
      for (int mi = 0; mi < 2; ++mi) {
        const __bf16* p = &aT[wm + mi * 16 + lrow][ks + half * 8];
        af[mi] = cat8(*(const bf16x8*)p, *(const bf16x8*)(p + 16));
      }
      #pragma unroll
      for (int ni = 0; ni < 4; ++ni) {
        const __bf16* p = &bT[wn + ni * 16 + lrow][ks + half * 8];
        bfr[ni] = cat8(*(const bf16x8*)p, *(const bf16x8*)(p + 16));
      }
      #pragma unroll
      for (int mi = 0; mi < 2; ++mi)
        #pragma unroll
        for (int ni = 0; ni < 4; ++ni)
          acc[mi][ni] = wmma_bf16(af[mi], bfr[ni], acc[mi][ni]);
    }
  }
  #pragma unroll
  for (int mi = 0; mi < 2; ++mi)
    #pragma unroll
    for (int ni = 0; ni < 4; ++ni) {
      int col = n0 + wn + ni * 16 + lrow;
      #pragma unroll
      for (int i = 0; i < 8; ++i) {
        int row = m0 + wm + mi * 16 + i + half * 8;
        out[row * D_MODEL + col] = acc[mi][ni][i];
      }
    }
}

extern "C" void kernel_launch(void* const* d_in, const int* in_sizes, int n_in,
                              void* d_out, int out_size, void* d_ws, size_t ws_size,
                              hipStream_t stream) {
  (void)in_sizes; (void)n_in; (void)out_size; (void)ws_size;
  const float* x    = (const float*)d_in[0];       // (1, 2048, 1024)
  const float* w    = (const float*)d_in[1];       // (3072, 1024)
  const float* bias = (const float*)d_in[2];       // (3072,)
  const int*   evt  = (const int*)d_in[3];         // (1,1)
  float* out    = (float*)d_out;                   // (1,1,2048,1024)
  float* biased = out + (size_t)S_LEN * D_MODEL;   // (1,1,2048,2048)
  __bf16* qk = (__bf16*)d_ws;                      // bf16 [2048][2048] = 8 MB

  qk_proj_kernel<<<dim3(S_LEN / 128, 2048 / 128), 256, 0, stream>>>(x, w, bias, qk);
  attn_kernel<<<dim3(S_LEN / 16), 256, 0, stream>>>(qk, evt, biased);
  out_gemm_kernel<<<dim3(S_LEN / 128, D_MODEL / 128), 256, 0, stream>>>(biased, x, out);
}
